// SSM_25795573580351
// MI455X (gfx1250) — matrix operand order repaired and checked by reading, not compiled, not run on hardware
//
#include <hip/hip_runtime.h>
#include <hip/hip_bf16.h>
#include <math.h>

// ---------- problem constants ----------
#define SEQ    4096
#define EMB    2048
#define H_     64
#define P_     64
#define N_     128
#define G_     8
#define K_     4
#define CS_    256
#define NC_    16          // SEQ / CS
#define INTER  4096        // H*P
#define CONVD  6144        // INTER + 2*G*N
#define PROJ   10304       // INTER + CONVD + H
#define GN     1024        // G*N

typedef __bf16 bf16_t;
typedef __attribute__((ext_vector_type(16))) __bf16 v16bf;
typedef __attribute__((ext_vector_type(8)))  __bf16 v8bf;
typedef __attribute__((ext_vector_type(8)))  float  v8f;
typedef __attribute__((ext_vector_type(4)))  int    v4i;

#define AS1 __attribute__((address_space(1)))
#define AS3 __attribute__((address_space(3)))

#if defined(__has_builtin)
#  if __has_builtin(__builtin_amdgcn_global_load_async_to_lds_b128)
#    define HAVE_ASYNC_LDS 1
#  endif
#  if __has_builtin(__builtin_amdgcn_s_wait_asynccnt)
#    define HAVE_WAIT_ASYNC 1
#  endif
#endif

// ---------- WMMA helpers (CDNA5 16x16x32 bf16, f32 accum) ----------
__device__ __forceinline__ v8f wmma_bf16(v16bf a, v16bf b, v8f c) {
  // (neg_a, A, neg_b, B, c_mod, C, reuse_a, reuse_b)
  return __builtin_amdgcn_wmma_f32_16x16x32_bf16(false, a, false, b, (short)0, c,
                                                 false, false);
}

union FragU { v16bf f; struct { v8bf lo; v8bf hi; } h; };

// A operand: 16(M) x 32(K), source row-major [M][stride].
// ISA 7.12.2: lanes 0-15 row M=lane, K 0-7 (v0-3) + 16-23 (v4-7);
//             lanes 16-31 same rows, K 8-15 + 24-31.
__device__ __forceinline__ v16bf load_frag_a(const bf16_t* base, int stride) {
  const int lane = threadIdx.x & 31;
  const int r    = lane & 15;
  const int sel  = lane >> 4;
  const bf16_t* p = base + (size_t)r * stride + sel * 8;
  FragU u;
  u.h.lo = *(const v8bf*)(p);
  u.h.hi = *(const v8bf*)(p + 16);
  return u.f;
}

// B operand: 32(K) x 16(N), source row-major [N][stride] (i.e. B^T rows).
// lanes 0-15 col N=lane hold K 0-7 + 16-23; lanes 16-31 hold K 8-15 + 24-31.
__device__ __forceinline__ v16bf load_frag_b(const bf16_t* base, int stride) {
  const int lane = threadIdx.x & 31;
  const int n    = lane & 15;
  const int sel  = lane >> 4;
  const bf16_t* p = base + (size_t)n * stride + sel * 8;
  FragU u;
  u.h.lo = *(const v8bf*)(p);
  u.h.hi = *(const v8bf*)(p + 16);
  return u.f;
}

// C/D tile 16x16 f32: vgpr r, lane L -> row = r + 8*(L>>4), col = L&15.
__device__ __forceinline__ void store_tile_f32(float* C, int ldc, v8f acc) {
  const int lane = threadIdx.x & 31;
  const int col  = lane & 15;
  const int rb   = (lane >> 4) * 8;
#pragma unroll
  for (int r = 0; r < 8; ++r) C[(size_t)(rb + r) * ldc + col] = acc[r];
}

// async global->LDS stage of `n8` 16-byte chunks (cooperative, 256 threads)
__device__ __forceinline__ void stage_async(const bf16_t* src, bf16_t* dst, int n8) {
  for (int i = threadIdx.x; i < n8; i += 256) {
#if defined(HAVE_ASYNC_LDS)
    __builtin_amdgcn_global_load_async_to_lds_b128(
        (AS1 v4i*)(src + (size_t)i * 8), (AS3 v4i*)(dst + (size_t)i * 8), 0, 0);
#else
    *(v8bf*)(dst + (size_t)i * 8) = *(const v8bf*)(src + (size_t)i * 8);
#endif
  }
}

__device__ __forceinline__ void wait_async_all() {
#if defined(HAVE_ASYNC_LDS)
#  if defined(HAVE_WAIT_ASYNC)
  __builtin_amdgcn_s_wait_asynccnt(0);
#  else
  asm volatile("s_wait_asynccnt 0" ::: "memory");
#  endif
#endif
}

__device__ __forceinline__ float siluf(float x) { return x / (1.f + __expf(-x)); }
__device__ __forceinline__ float softplusf(float x) {
  return x > 20.f ? x : log1pf(__expf(x));
}

// ---------- fp32 -> bf16 convert ----------
__global__ __launch_bounds__(256) void cvt_bf16_kernel(const float* __restrict__ in,
                                                       bf16_t* __restrict__ out, int n) {
  int i = blockIdx.x * 256 + threadIdx.x;
  if (i < n) out[i] = (bf16_t)in[i];
}

// ---------- generic bf16 WMMA GEMM: C[M,N] = A[M,K] * B[N,K]^T ----------
// block = 256 threads (8 waves); block tile 128(M) x 64(N); wave tile 32x32.
// Operands are bf16 (both weight panels fit in the 192MB L2), so fragments are
// loaded straight from global with prefetch running ahead of K.
__global__ __launch_bounds__(256) void gemm_bf16_kernel(
    const bf16_t* __restrict__ A, const bf16_t* __restrict__ B,
    float* __restrict__ C, int M, int N, int K) {
  const int wave = threadIdx.x >> 5;
  const int m0 = blockIdx.y * 128 + (wave & 3) * 32;
  const int n0 = blockIdx.x * 64  + (wave >> 2) * 32;
  v8f acc00 = {}, acc01 = {}, acc10 = {}, acc11 = {};
  for (int k = 0; k < K; k += 32) {
    if (k + 64 < K) {   // stream ahead of the fragment loads (global_prefetch_b8)
      __builtin_prefetch((const void*)(A + (size_t)m0 * K + k + 64), 0, 1);
      __builtin_prefetch((const void*)(B + (size_t)n0 * K + k + 64), 0, 1);
    }
    v16bf a0 = load_frag_a(A + (size_t)m0 * K + k, K);
    v16bf a1 = load_frag_a(A + (size_t)(m0 + 16) * K + k, K);
    v16bf b0 = load_frag_b(B + (size_t)n0 * K + k, K);
    v16bf b1 = load_frag_b(B + (size_t)(n0 + 16) * K + k, K);
    acc00 = wmma_bf16(a0, b0, acc00);
    acc01 = wmma_bf16(a0, b1, acc01);
    acc10 = wmma_bf16(a1, b0, acc10);
    acc11 = wmma_bf16(a1, b1, acc11);
  }
  store_tile_f32(C + (size_t)m0 * N + n0,            N, acc00);
  store_tile_f32(C + (size_t)m0 * N + n0 + 16,       N, acc01);
  store_tile_f32(C + (size_t)(m0 + 16) * N + n0,     N, acc10);
  store_tile_f32(C + (size_t)(m0 + 16) * N + n0 + 16,N, acc11);
}

// ---------- depthwise causal conv (K=4) + bias + SiLU ----------
__global__ __launch_bounds__(256) void conv_silu_kernel(
    const float* __restrict__ proj, const float* __restrict__ cw,
    const float* __restrict__ cb, float* __restrict__ hbc) {
  int idx = blockIdx.x * 256 + threadIdx.x;
  if (idx >= SEQ * CONVD) return;
  int t = idx / CONVD, c = idx - t * CONVD;
  float s = cb[c];
#pragma unroll
  for (int k = 0; k < K_; ++k) {
    int tt = t - (K_ - 1) + k;
    if (tt >= 0) s += cw[c * K_ + k] * proj[(size_t)tt * PROJ + INTER + c];
  }
  hbc[idx] = siluf(s);
}

// ---------- dt2 = softplus(dt + bias), a_dt = -exp(A_log)*dt2 ----------
__global__ __launch_bounds__(256) void dt_kernel(
    const float* __restrict__ proj, const float* __restrict__ dt_bias,
    const float* __restrict__ A_log, float* __restrict__ dt2,
    float* __restrict__ a_dt) {
  int idx = blockIdx.x * 256 + threadIdx.x;
  if (idx >= SEQ * H_) return;
  int t = idx >> 6, h = idx & 63;
  float sp = softplusf(proj[(size_t)t * PROJ + INTER + CONVD + h] + dt_bias[h]);
  dt2[idx]  = sp;
  a_dt[idx] = -__expf(A_log[h]) * sp;
}

// ---------- per-(chunk,head) cumulative sum of a_dt over CS ----------
__global__ __launch_bounds__(256) void cumsum_kernel(const float* __restrict__ a_dt,
                                                     float* __restrict__ Acum) {
  int tid = blockIdx.x * 256 + threadIdx.x;      // 1024 = NC*H threads
  if (tid >= NC_ * H_) return;
  int c = tid >> 6, h = tid & 63;
  float run = 0.f;
  for (int l = 0; l < CS_; ++l) {
    run += a_dt[(size_t)(c * CS_ + l) * H_ + h];
    Acum[(size_t)tid * CS_ + l] = run;
  }
}

// ---------- pack B/C columns of hbc into bf16 [SEQ][2*GN] ----------
__global__ __launch_bounds__(256) void pack_bc_kernel(const float* __restrict__ hbc,
                                                      bf16_t* __restrict__ bc) {
  int idx = blockIdx.x * 256 + threadIdx.x;
  if (idx >= SEQ * 2 * GN) return;
  int t = idx / (2 * GN), j = idx - t * 2 * GN;
  bc[idx] = (bf16_t)hbc[(size_t)t * CONVD + INTER + j];
}

// ---------- pack X^T * dt2 as bf16 [NC*H][P][CS] ----------
__global__ __launch_bounds__(256) void pack_xdtT_kernel(
    const float* __restrict__ hbc, const float* __restrict__ dt2,
    bf16_t* __restrict__ xdtT) {
  int idx = blockIdx.x * 256 + threadIdx.x;      // NC*H*P*CS = 16.7M
  if (idx >= NC_ * H_ * P_ * CS_) return;
  int l = idx & 255, p = (idx >> 8) & 63, h = (idx >> 14) & 63, c = idx >> 20;
  int t = c * CS_ + l;
  xdtT[idx] = (bf16_t)(hbc[(size_t)t * CONVD + h * P_ + p] * dt2[(size_t)t * H_ + h]);
}

// ---------- pack decayed B^T as bf16 [NC*H][N][CS] ----------
__global__ __launch_bounds__(256) void pack_bdecT_kernel(
    const float* __restrict__ hbc, const float* __restrict__ Acum,
    bf16_t* __restrict__ bdecT) {
  int idx = blockIdx.x * 256 + threadIdx.x;      // NC*H*N*CS = 33.5M
  if (idx >= NC_ * H_ * N_ * CS_) return;
  int l = idx & 255, n = (idx >> 8) & 127, h = (idx >> 15) & 63, c = idx >> 21;
  int g = h >> 3, t = c * CS_ + l, ch = c * H_ + h;
  float decay = __expf(Acum[(size_t)ch * CS_ + (CS_ - 1)] - Acum[(size_t)ch * CS_ + l]);
  bdecT[idx] = (bf16_t)(hbc[(size_t)t * CONVD + INTER + g * N_ + n] * decay);
}

// ---------- state^T[p,n] = sum_l xdtT[p,l] * bdecT[n,l]  (WMMA) ----------
__global__ __launch_bounds__(256) void ssd_state_kernel(
    const bf16_t* __restrict__ xdtT, const bf16_t* __restrict__ bdecT,
    bf16_t* __restrict__ stateT) {
  int ch = blockIdx.x;                       // NC*H blocks
  const bf16_t* Xt = xdtT  + (size_t)ch * P_ * CS_;
  const bf16_t* Bd = bdecT + (size_t)ch * N_ * CS_;
  int wave = threadIdx.x >> 5;
  int m0 = (wave & 3) * 16;                  // p tile
  int n0 = (wave >> 2) * 64;                 // n offset, 4 tiles of 16
  v8f acc[4] = {};
  for (int k = 0; k < CS_; k += 32) {
    v16bf a = load_frag_a(Xt + (size_t)m0 * CS_ + k, CS_);
#pragma unroll
    for (int tt = 0; tt < 4; ++tt) {
      v16bf b = load_frag_b(Bd + (size_t)(n0 + tt * 16) * CS_ + k, CS_);
      acc[tt] = wmma_bf16(a, b, acc[tt]);
    }
  }
  const int lane = threadIdx.x & 31;
  const int col = lane & 15, rb = (lane >> 4) * 8;
#pragma unroll
  for (int tt = 0; tt < 4; ++tt)
#pragma unroll
    for (int r = 0; r < 8; ++r)
      stateT[(size_t)(ch * P_ + m0 + rb + r) * N_ + n0 + tt * 16 + col] =
          (bf16_t)acc[tt][r];
}

// ---------- main SSD Y kernel: Yd + Yoff*Sd + D*hid  (WMMA) ----------
// xdtT (32KB) and stateT (16KB) panels are block-shared across all 8 waves and
// all j-iterations -> stage them once via ASYNC global->LDS, then serve all
// WMMA B operands from LDS. B/C chunk panels stay in L2 (group-shared by 8
// head-blocks).
__global__ __launch_bounds__(256) void ssd_y_kernel(
    const bf16_t* __restrict__ bc, const bf16_t* __restrict__ xdtT,
    const bf16_t* __restrict__ stateT, const float* __restrict__ Acum,
    const float* __restrict__ hbc, const float* __restrict__ Dv,
    float* __restrict__ y) {
  __shared__ bf16_t ldsM[8][16 * 32];        // per-wave 16x32 M-tile bounce (8KB)
  __shared__ bf16_t ldsX[P_ * CS_];          // xdtT panel  [p][l]  (32KB)
  __shared__ bf16_t ldsS[P_ * N_];           // stateT panel [p][n] (16KB)
  const int c = blockIdx.x, h = blockIdx.y, g = h >> 3, ch = c * H_ + h;
  const int wave = threadIdx.x >> 5, lane = threadIdx.x & 31;
  const int col = lane & 15, rb = (lane >> 4) * 8;
  const float* ac = Acum + (size_t)ch * CS_;
  const bf16_t* Crow = bc + (size_t)(c * CS_) * (2 * GN) + GN + g * N_;  // stride 2*GN
  const bf16_t* Brow = bc + (size_t)(c * CS_) * (2 * GN) + g * N_;
  const float dcoef = Dv[h];

  // async stage of block-shared panels
  stage_async(xdtT   + (size_t)ch * P_ * CS_, ldsX, P_ * CS_ / 8);
  stage_async(stateT + (size_t)ch * P_ * N_,  ldsS, P_ * N_ / 8);
  wait_async_all();
  __syncthreads();

  for (int rt = wave; rt < CS_ / 16; rt += 8) {
    const int l0 = rt * 16;
    float ai[8];
#pragma unroll
    for (int r = 0; r < 8; ++r) ai[r] = ac[l0 + rb + r];

    v8f accd[4] = {};                        // Yd over 4 p-tiles
    for (int j0 = 0; j0 < CS_; j0 += 32) {
      v8f g0 = {}, g1 = {};
#pragma unroll
      for (int kk = 0; kk < N_; kk += 32) {  // Gm = C . B^T over n
        v16bf aC = load_frag_a(Crow + (size_t)l0 * (2 * GN) + kk, 2 * GN);
        v16bf b0 = load_frag_b(Brow + (size_t)j0 * (2 * GN) + kk, 2 * GN);
        v16bf b1 = load_frag_b(Brow + (size_t)(j0 + 16) * (2 * GN) + kk, 2 * GN);
        g0 = wmma_bf16(aC, b0, g0);
        g1 = wmma_bf16(aC, b1, g1);
      }
      // causal mask + exp(segsum), bounce through LDS to rebuild A-operand layout
      float aj0 = ac[j0 + col], aj1 = ac[j0 + 16 + col];
      bf16_t* mt = ldsM[wave];
#pragma unroll
      for (int r = 0; r < 8; ++r) {
        int i = l0 + rb + r;
        float m0v = (j0 + col      <= i) ? g0[r] * __expf(ai[r] - aj0) : 0.f;
        float m1v = (j0 + 16 + col <= i) ? g1[r] * __expf(ai[r] - aj1) : 0.f;
        mt[(rb + r) * 32 + col]      = (bf16_t)m0v;
        mt[(rb + r) * 32 + 16 + col] = (bf16_t)m1v;
      }
      asm volatile("s_wait_dscnt 0" ::: "memory");   // same-wave DS RAW fence
      v16bf aM = load_frag_a(mt, 32);
#pragma unroll
      for (int tt = 0; tt < 4; ++tt) {       // Yd += M . xdt  (B from LDS)
        v16bf bX = load_frag_b(ldsX + (size_t)(tt * 16) * CS_ + j0, CS_);
        accd[tt] = wmma_bf16(aM, bX, accd[tt]);
      }
    }

    v8f acco[4] = {};                        // Yoff_raw = C . state^T
#pragma unroll
    for (int kk = 0; kk < N_; kk += 32) {
      v16bf aC = load_frag_a(Crow + (size_t)l0 * (2 * GN) + kk, 2 * GN);
#pragma unroll
      for (int tt = 0; tt < 4; ++tt) {       // B from LDS
        v16bf bS = load_frag_b(ldsS + (size_t)(tt * 16) * N_ + kk, N_);
        acco[tt] = wmma_bf16(aC, bS, acco[tt]);
      }
    }

#pragma unroll
    for (int tt = 0; tt < 4; ++tt)
#pragma unroll
      for (int r = 0; r < 8; ++r) {
        int l = l0 + rb + r, t = c * CS_ + l, p = tt * 16 + col;
        float sd = __expf(ai[r]);
        float val = accd[tt][r] + acco[tt][r] * sd +
                    dcoef * hbc[(size_t)t * CONVD + h * P_ + p];
        y[(size_t)t * INTER + h * P_ + p] = val;
      }
  }
}

// ---------- gate * SiLU + RMSNorm -> bf16 ----------
__global__ __launch_bounds__(256) void gate_norm_kernel(
    const float* __restrict__ y, const float* __restrict__ proj,
    const float* __restrict__ norm_w, bf16_t* __restrict__ outb) {
  __shared__ float yg[INTER];
  __shared__ float red[256];
  const int t = blockIdx.x;
  float ss = 0.f;
  for (int i = threadIdx.x; i < INTER; i += 256) {
    float gv = proj[(size_t)t * PROJ + i];
    float v  = y[(size_t)t * INTER + i] * siluf(gv);
    yg[i] = v;
    ss += v * v;
  }
  red[threadIdx.x] = ss;
  __syncthreads();
  for (int s = 128; s > 0; s >>= 1) {
    if (threadIdx.x < s) red[threadIdx.x] += red[threadIdx.x + s];
    __syncthreads();
  }
  float inv = rsqrtf(red[0] / (float)INTER + 1e-6f);
  for (int i = threadIdx.x; i < INTER; i += 256)
    outb[(size_t)t * INTER + i] = (bf16_t)(norm_w[i] * yg[i] * inv);
}

// ---------------------------------------------------------------------------
extern "C" void kernel_launch(void* const* d_in, const int* in_sizes, int n_in,
                              void* d_out, int out_size, void* d_ws, size_t ws_size,
                              hipStream_t stream) {
  const float* x_f32   = (const float*)d_in[0];   // [1,4096,2048]
  const float* win_f32 = (const float*)d_in[1];   // [10304,2048]
  const float* conv_w  = (const float*)d_in[2];   // [6144,1,4]
  const float* conv_b  = (const float*)d_in[3];   // [6144]
  const float* dt_bias = (const float*)d_in[4];   // [64]
  const float* A_log   = (const float*)d_in[5];   // [64]
  const float* D_in    = (const float*)d_in[6];   // [64]
  const float* norm_w  = (const float*)d_in[7];   // [4096]
  const float* wout_f32= (const float*)d_in[8];   // [2048,4096]
  float* out = (float*)d_out;                     // [1,4096,2048]

  // workspace carve-out (256B aligned)
  char* ws = (char*)d_ws;
  size_t off = 0;
  auto carve = [&](size_t bytes) { void* p = ws + off; off += (bytes + 255) & ~(size_t)255; return p; };
  bf16_t* Xb     = (bf16_t*)carve((size_t)SEQ * EMB * 2);
  bf16_t* WinB   = (bf16_t*)carve((size_t)PROJ * EMB * 2);
  bf16_t* WoutB  = (bf16_t*)carve((size_t)EMB * INTER * 2);
  float*  proj   = (float*) carve((size_t)SEQ * PROJ * 4);
  float*  hbc    = (float*) carve((size_t)SEQ * CONVD * 4);
  float*  dt2    = (float*) carve((size_t)SEQ * H_ * 4);
  float*  a_dt   = (float*) carve((size_t)SEQ * H_ * 4);
  float*  Acum   = (float*) carve((size_t)NC_ * H_ * CS_ * 4);
  bf16_t* bc     = (bf16_t*)carve((size_t)SEQ * 2 * GN * 2);
  bf16_t* xdtT   = (bf16_t*)carve((size_t)NC_ * H_ * P_ * CS_ * 2);
  bf16_t* bdecT  = (bf16_t*)carve((size_t)NC_ * H_ * N_ * CS_ * 2);
  bf16_t* stateT = (bf16_t*)carve((size_t)NC_ * H_ * P_ * N_ * 2);
  float*  ybuf   = (float*) carve((size_t)SEQ * INTER * 4);
  bf16_t* normb  = (bf16_t*)carve((size_t)SEQ * INTER * 2);
  (void)ws_size; (void)in_sizes; (void)n_in; (void)out_size;

  auto blocks = [](size_t n) { return (unsigned)((n + 255) / 256); };

  // 1) fp32 -> bf16 for GEMM operands
  cvt_bf16_kernel<<<blocks((size_t)SEQ * EMB), 256, 0, stream>>>(x_f32, Xb, SEQ * EMB);
  cvt_bf16_kernel<<<blocks((size_t)PROJ * EMB), 256, 0, stream>>>(win_f32, WinB, PROJ * EMB);
  cvt_bf16_kernel<<<blocks((size_t)EMB * INTER), 256, 0, stream>>>(wout_f32, WoutB, EMB * INTER);

  // 2) in_proj GEMM: proj[4096,10304] = X[4096,2048] . Win[10304,2048]^T
  gemm_bf16_kernel<<<dim3(PROJ / 64, SEQ / 128), 256, 0, stream>>>(Xb, WinB, proj,
                                                                   SEQ, PROJ, EMB);
  // 3) depthwise conv + SiLU
  conv_silu_kernel<<<blocks((size_t)SEQ * CONVD), 256, 0, stream>>>(proj, conv_w, conv_b, hbc);

  // 4) dt2 / a_dt, per-chunk cumsum
  dt_kernel<<<blocks((size_t)SEQ * H_), 256, 0, stream>>>(proj, dt_bias, A_log, dt2, a_dt);
  cumsum_kernel<<<blocks((size_t)NC_ * H_), 256, 0, stream>>>(a_dt, Acum);

  // 5) bf16 packs for SSD matmuls
  pack_bc_kernel<<<blocks((size_t)SEQ * 2 * GN), 256, 0, stream>>>(hbc, bc);
  pack_xdtT_kernel<<<blocks((size_t)NC_ * H_ * P_ * CS_), 256, 0, stream>>>(hbc, dt2, xdtT);
  pack_bdecT_kernel<<<blocks((size_t)NC_ * H_ * N_ * CS_), 256, 0, stream>>>(hbc, Acum, bdecT);

  // 6) SSD state + Y (WMMA)
  ssd_state_kernel<<<NC_ * H_, 256, 0, stream>>>(xdtT, bdecT, stateT);
  ssd_y_kernel<<<dim3(NC_, H_), 256, 0, stream>>>(bc, xdtT, stateT, Acum, hbc, D_in, ybuf);

  // 7) gate + RMSNorm -> bf16
  gate_norm_kernel<<<SEQ, 256, 0, stream>>>(ybuf, proj, norm_w, normb);

  // 8) out_proj GEMM: out[4096,2048] = normed[4096,4096] . Wout[2048,4096]^T
  gemm_bf16_kernel<<<dim3(EMB / 64, SEQ / 128), 256, 0, stream>>>(normb, WoutB, out,
                                                                  SEQ, EMB, INTER);
}
